// ExactSphericalKernelCausalAttention_56367150793419
// MI455X (gfx1250) — matrix operand order repaired and checked by reading, not compiled, not run on hardware
//
#include <hip/hip_runtime.h>
#include <hip/hip_bf16.h>

// ---------------------------------------------------------------------------
// ExactSphericalKernelCausalAttention for MI455X (gfx1250, wave32, WMMA)
// B=2, T=2048, E=1024, H=16, D=64
// ---------------------------------------------------------------------------

typedef __bf16 bf16;
typedef bf16  v16bf __attribute__((ext_vector_type(16)));
typedef bf16  v8bf  __attribute__((ext_vector_type(8)));
typedef float v8f   __attribute__((ext_vector_type(8)));

#define WMMA_BF16(a, b, c) \
  __builtin_amdgcn_wmma_f32_16x16x32_bf16(false, (a), false, (b), (short)0, (c), false, false)

static constexpr int Bsz = 2;
static constexpr int T   = 2048;
static constexpr int E   = 1024;
static constexpr int H   = 16;
static constexpr int D   = 64;

// ---- WMMA operand loaders (bf16, 16x16x32) --------------------------------
// A (16xK=32): lane l -> row = l&15 ; element e -> K = 16*(e>>3) + 8*(l>>4) + (e&7)
static __device__ __forceinline__ v16bf load_a_tile(const bf16* __restrict__ A, int lda,
                                                    int mbase, int kbase, int lane) {
  int row = lane & 15, half = lane >> 4;
  const bf16* p0 = A + (size_t)(mbase + row) * lda + kbase + 8 * half;
  v8bf lo = *(const v8bf*)p0;
  v8bf hi = *(const v8bf*)(p0 + 16);
  v16bf r;
#pragma unroll
  for (int i = 0; i < 8; ++i) { r[i] = lo[i]; r[i + 8] = hi[i]; }
  return r;
}

// B (K=32 x 16): lane l -> col = l&15 ; element e -> K = 16*(e>>3) + 8*(l>>4) + (e&7)
// Bw is stored row-major as [N, K] (i.e. each "column" of B is a contiguous row).
static __device__ __forceinline__ v16bf load_b_tile(const bf16* __restrict__ Bw, int ldb,
                                                    int nbase, int kbase, int lane) {
  int col = lane & 15, half = lane >> 4;
  const bf16* p = Bw + (size_t)(nbase + col) * ldb + kbase + 8 * half;
  v8bf lo = *(const v8bf*)p;
  v8bf hi = *(const v8bf*)(p + 16);
  v16bf r;
#pragma unroll
  for (int i = 0; i < 8; ++i) { r[i] = lo[i]; r[i + 8] = hi[i]; }
  return r;
}

// ---- elementwise f32 -> bf16 ----------------------------------------------
__global__ void cvt_f32_bf16(const float* __restrict__ in, bf16* __restrict__ out, int n) {
  int i = blockIdx.x * 256 + threadIdx.x;
  if (i < n) out[i] = (bf16)in[i];
}

// ---- C[M,N] = A[M,K](bf16) @ Bw[N,K]^T(bf16) + bias[N], fp32 out ----------
// one wave per block; wave computes a 32x64 tile (2 M-subtiles x 4 N-subtiles).
// Software-pipelined: loads for iteration k+1 are issued before the 8 WMMAs of
// iteration k, so the single s_wait_loadcnt lands a full WMMA-burst later.
__global__ void __launch_bounds__(32)
gemm_bf16_f32(const bf16* __restrict__ A, const bf16* __restrict__ Bw,
              const float* __restrict__ bias, float* __restrict__ C,
              int M, int N, int K) {
  int lane  = threadIdx.x;
  int nbase = blockIdx.x * 64;
  int mbase = blockIdx.y * 32;
  v8f acc[8] = {};   // acc[mi*4 + t]

  v16bf a0 = load_a_tile(A, K, mbase,      0, lane);
  v16bf a1 = load_a_tile(A, K, mbase + 16, 0, lane);
  v16bf b0 = load_b_tile(Bw, K, nbase +  0, 0, lane);
  v16bf b1 = load_b_tile(Bw, K, nbase + 16, 0, lane);
  v16bf b2 = load_b_tile(Bw, K, nbase + 32, 0, lane);
  v16bf b3 = load_b_tile(Bw, K, nbase + 48, 0, lane);

  for (int kb = 0; kb < K; kb += 32) {
    int nk = (kb + 32 < K) ? kb + 32 : 0;   // wrap on last iter (redundant, harmless)
    v16bf na0 = load_a_tile(A, K, mbase,      nk, lane);
    v16bf na1 = load_a_tile(A, K, mbase + 16, nk, lane);
    v16bf nb0 = load_b_tile(Bw, K, nbase +  0, nk, lane);
    v16bf nb1 = load_b_tile(Bw, K, nbase + 16, nk, lane);
    v16bf nb2 = load_b_tile(Bw, K, nbase + 32, nk, lane);
    v16bf nb3 = load_b_tile(Bw, K, nbase + 48, nk, lane);

    acc[0] = WMMA_BF16(a0, b0, acc[0]);
    acc[1] = WMMA_BF16(a0, b1, acc[1]);
    acc[2] = WMMA_BF16(a0, b2, acc[2]);
    acc[3] = WMMA_BF16(a0, b3, acc[3]);
    acc[4] = WMMA_BF16(a1, b0, acc[4]);
    acc[5] = WMMA_BF16(a1, b1, acc[5]);
    acc[6] = WMMA_BF16(a1, b2, acc[6]);
    acc[7] = WMMA_BF16(a1, b3, acc[7]);

    a0 = na0; a1 = na1; b0 = nb0; b1 = nb1; b2 = nb2; b3 = nb3;
  }

  int col = lane & 15, half = lane >> 4;
#pragma unroll
  for (int mi = 0; mi < 2; ++mi) {
#pragma unroll
    for (int t = 0; t < 4; ++t) {
      int n = nbase + 16 * t + col;
      float bv = bias[n];
#pragma unroll
      for (int j = 0; j < 8; ++j) {
        int m = mbase + 16 * mi + j + 8 * half;
        C[(size_t)m * N + n] = acc[mi * 4 + t][j] + bv;
      }
    }
  }
}

// ---- normalize q,k rows; repack to [B,H,T,D] bf16 ; v -> [B,H,D,T] bf16 ---
__global__ void pack_norm(const float* __restrict__ qkv,
                          bf16* __restrict__ qn, bf16* __restrict__ kn,
                          bf16* __restrict__ vT) {
  int r = blockIdx.x * blockDim.x + threadIdx.x;   // r = ((b*T)+t)*H + h
  int h = r & 15;
  int t = (r >> 4) & (T - 1);
  int b = r >> 15;
  const float* base = qkv + (size_t)(b * T + t) * (3 * E) + h * D;
  float sq = 0.f, sk = 0.f;
  for (int d = 0; d < D; ++d) {
    float q = base[d];       sq += q * q;
    float k = base[E + d];   sk += k * k;
  }
  float iq = 1.f / fmaxf(sqrtf(sq), 1e-12f);
  float ik = 1.f / fmaxf(sqrtf(sk), 1e-12f);
  size_t o = ((size_t)(b * H + h) * T + t) * D;
  for (int d = 0; d < D; ++d) {
    qn[o + d] = (bf16)(base[d] * iq);
    kn[o + d] = (bf16)(base[E + d] * ik);
  }
  size_t vo = (size_t)(b * H + h) * D * T + t;
  for (int d = 0; d < D; ++d)
    vT[vo + (size_t)d * T] = (bf16)base[2 * E + d];
}

// ---- streaming causal "spherical kernel" attention -------------------------
// grid = (T/16, B*H), one wave per 16-query tile.
// scores:  x_dot = qn @ kn^T   (WMMA, K-dim = D = 64)
// Kmat   = x_dot^2 / max(2+eps - 2*x_dot, 1e-6), causal-masked  (fp32 VALU)
// numer  = Kmat @ v            (WMMA, K-dim = keys, via LDS relayout)
// z      = Kmat @ ones-col     (extra WMMA, lands in column 0)
__global__ void __launch_bounds__(32)
spherical_attn(const bf16* __restrict__ qn, const bf16* __restrict__ kn,
               const bf16* __restrict__ vT, bf16* __restrict__ attnb) {
  __shared__ __attribute__((aligned(16))) bf16 sK[16 * 32];
  __shared__ float sz[16];

  int lane  = threadIdx.x;
  int qt    = blockIdx.x;
  int bh    = blockIdx.y;
  int b     = bh >> 4;
  int h     = bh & 15;
  int qbase = qt * 16;
  int row   = lane & 15, half = lane >> 4, col = lane & 15;

  const bf16* qp = qn + (size_t)bh * T * D;
  const bf16* kp = kn + (size_t)bh * T * D;
  const bf16* vp = vT + (size_t)bh * D * T;

  // query A-tiles for the full D=64 contraction (reused for every key tile)
  v16bf qa0 = load_a_tile(qp, D, qbase, 0,  lane);
  v16bf qa1 = load_a_tile(qp, D, qbase, 32, lane);

  // ones B-tile: column 0 == 1.0 -> z accumulates as a matmul column
  bf16 onev = (bf16)((col == 0) ? 1.0f : 0.0f);
  v16bf onesb;
#pragma unroll
  for (int i = 0; i < 16; ++i) onesb[i] = onev;

  v8f numer[4] = {};
  v8f zacc = {};

  const float Cc = 2.0f + 1e-6f;
  int nkb = (qbase + 16 + 31) >> 5;          // 32-key blocks covering [0, qbase+16)

  for (int kb2 = 0; kb2 < nkb; ++kb2) {
    int kb = kb2 * 32;
    // --- scores for two 16-key tiles, transform, stage into LDS -------------
#pragma unroll
    for (int s = 0; s < 2; ++s) {
      int kt = kb + 16 * s;
      v8f sc = {};
      v16bf b0 = load_b_tile(kp, D, kt, 0,  lane);
      v16bf b1 = load_b_tile(kp, D, kt, 32, lane);
      sc = WMMA_BF16(qa0, b0, sc);
      sc = WMMA_BF16(qa1, b1, sc);
      int kg = kt + col;                      // this lane's global key index
#pragma unroll
      for (int j = 0; j < 8; ++j) {
        int qg = qbase + j + 8 * half;
        float x  = sc[j];
        float dn = fmaxf(Cc - 2.0f * x, 1e-6f);
        float Kv = (kg <= qg) ? (x * x / dn) : 0.0f;
        sK[(j + 8 * half) * 32 + 16 * s + col] = (bf16)Kv;
      }
    }
    __syncthreads();
    // --- reload Kmat in A-operand layout (16 queries x 32 keys) -------------
    v16bf ka;
    {
      const v8bf* p0 = (const v8bf*)&sK[row * 32 + 8 * half];
      const v8bf* p1 = (const v8bf*)&sK[row * 32 + 8 * half + 16];
      v8bf lo = *p0, hi = *p1;
#pragma unroll
      for (int i = 0; i < 8; ++i) { ka[i] = lo[i]; ka[i + 8] = hi[i]; }
    }
    // --- numer += Kmat @ v ; z += Kmat @ ones -------------------------------
#pragma unroll
    for (int t = 0; t < 4; ++t) {
      v16bf vb = load_b_tile(vp, T, 16 * t, kb, lane);
      numer[t] = WMMA_BF16(ka, vb, numer[t]);
    }
    zacc = WMMA_BF16(ka, onesb, zacc);
    __syncthreads();
  }

  // broadcast z (column 0) to all lanes via LDS
  if (col == 0) {
#pragma unroll
    for (int j = 0; j < 8; ++j) sz[j + 8 * half] = zacc[j];
  }
  __syncthreads();

#pragma unroll
  for (int t = 0; t < 4; ++t) {
#pragma unroll
    for (int j = 0; j < 8; ++j) {
      int qg = qbase + j + 8 * half;
      float z = fmaxf(sz[j + 8 * half], 1e-6f);
      float o = numer[t][j] / z;
      attnb[((size_t)b * T + qg) * E + h * D + 16 * t + col] = (bf16)o;
    }
  }
}

// ---------------------------------------------------------------------------
extern "C" void kernel_launch(void* const* d_in, const int* in_sizes, int n_in,
                              void* d_out, int out_size, void* d_ws, size_t ws_size,
                              hipStream_t stream) {
  (void)in_sizes; (void)n_in; (void)out_size; (void)ws_size;
  const float* x     = (const float*)d_in[0];
  const float* w_qkv = (const float*)d_in[1];
  const float* b_qkv = (const float*)d_in[2];
  const float* w_out = (const float*)d_in[3];
  const float* b_out = (const float*)d_in[4];
  float* out = (float*)d_out;

  char* ws = (char*)d_ws;
  size_t off = 0;
  auto alloc = [&](size_t bytes) -> void* {
    void* p = ws + off;
    off += (bytes + 255) & ~(size_t)255;
    return p;
  };

  const size_t MT = (size_t)Bsz * T;                 // 4096 rows
  bf16*  xb    = (bf16*)alloc(MT * E * 2);           // x in bf16
  bf16*  wqb   = (bf16*)alloc((size_t)3 * E * E * 2);// w_qkv in bf16
  bf16*  wob   = (bf16*)alloc((size_t)E * E * 2);    // w_out in bf16
  float* qkv   = (float*)alloc(MT * 3 * E * 4);      // fp32 qkv
  bf16*  qnb   = (bf16*)alloc(MT * E * 2);           // [B,H,T,D]
  bf16*  knb   = (bf16*)alloc(MT * E * 2);           // [B,H,T,D]
  bf16*  vTb   = (bf16*)alloc(MT * E * 2);           // [B,H,D,T]
  bf16*  attnb = (bf16*)alloc(MT * E * 2);           // [B,T,E]

  // 1) bf16 conversions
  cvt_f32_bf16<<<(MT * E) / 256, 256, 0, stream>>>(x, xb, (int)(MT * E));
  cvt_f32_bf16<<<(3 * E * E) / 256, 256, 0, stream>>>(w_qkv, wqb, 3 * E * E);
  cvt_f32_bf16<<<(E * E) / 256, 256, 0, stream>>>(w_out, wob, E * E);

  // 2) qkv = x @ w_qkv^T + b_qkv        [4096 x 3072]
  gemm_bf16_f32<<<dim3((3 * E) / 64, MT / 32), 32, 0, stream>>>(
      xb, wqb, b_qkv, qkv, (int)MT, 3 * E, E);

  // 3) normalize + repack
  pack_norm<<<(Bsz * T * H) / 256, 256, 0, stream>>>(qkv, qnb, knb, vTb);

  // 4) causal spherical-kernel attention -> attnb [B,T,E] bf16
  spherical_attn<<<dim3(T / 16, Bsz * H), 32, 0, stream>>>(qnb, knb, vTb, attnb);

  // 5) out = attn @ w_out^T + b_out     [4096 x 1024] fp32
  gemm_bf16_f32<<<dim3(E / 64, MT / 32), 32, 0, stream>>>(
      attnb, wob, b_out, out, (int)MT, E, E);
}
